// GraphSAGE_33990371181412
// MI455X (gfx1250) — matrix operand order repaired and checked by reading, not compiled, not run on hardware
//
#include <hip/hip_runtime.h>

typedef __attribute__((ext_vector_type(16))) __bf16 v16bf;
typedef __attribute__((ext_vector_type(8)))  float  v8f;
typedef __attribute__((ext_vector_type(4)))  unsigned int v4u;
typedef __attribute__((ext_vector_type(8)))  int    v8i;
typedef __attribute__((ext_vector_type(4)))  int    v4i;

#define FEAT 128  // IN_DIM == HID_DIM == 128, K = 2*FEAT = 256

__device__ __forceinline__ __bf16 f2bf(float f) {
  unsigned u = __builtin_bit_cast(unsigned, f);
  u += 0x7FFFu + ((u >> 16) & 1u);           // round-to-nearest-even
  unsigned short h = (unsigned short)(u >> 16);
  return __builtin_bit_cast(__bf16, h);
}

// ---------------------------------------------------------------- utilities
__global__ void zero_f32(float* __restrict__ p, long long n4) {
  long long i = (long long)blockIdx.x * blockDim.x + threadIdx.x;
  long long stride = (long long)gridDim.x * blockDim.x;
  float4 z = make_float4(0.f, 0.f, 0.f, 0.f);
  for (; i < n4; i += stride) ((float4*)p)[i] = z;
}

__global__ void deg_count(const long long* __restrict__ dst,
                          float* __restrict__ deg, int n_edges) {
  int e = blockIdx.x * blockDim.x + threadIdx.x;
  if (e < n_edges) atomicAdd(&deg[(int)dst[e]], 1.0f);
}

__global__ void deg_recip(float* __restrict__ deg, int n) {
  int i = blockIdx.x * blockDim.x + threadIdx.x;
  if (i < n) deg[i] = 1.0f / fmaxf(deg[i], 1.0f);
}

// Pack Wl (FEAT x dout) on top of Wr (FEAT x dout) transposed into
// Wt[col][k], k in [0,256): k<FEAT -> Wl[k][col], else Wr[k-FEAT][col].
__global__ void prep_w(const float* __restrict__ Wl, const float* __restrict__ Wr,
                       __bf16* __restrict__ Wt, int dout) {
  int i = blockIdx.x * blockDim.x + threadIdx.x;
  if (i >= dout * 256) return;
  int col = i >> 8, k = i & 255;
  float v = (k < FEAT) ? Wl[k * dout + col] : Wr[(k - FEAT) * dout + col];
  Wt[i] = f2bf(v);
}

// ---------------------------------------------------------- edge scatter-add
// One thread per (edge, 4-feature chunk): float4 gather + 4 L2 atomics.
__global__ void scatter_feat(const float* __restrict__ x,
                             const long long* __restrict__ src,
                             const long long* __restrict__ dst,
                             float* __restrict__ summed, int n_edges) {
  long long idx = (long long)blockIdx.x * blockDim.x + threadIdx.x;
  int e = (int)(idx >> 5);
  if (e >= n_edges) return;
  int c = ((int)idx & 31) << 2;
  int s = (int)src[e], d = (int)dst[e];
  float4 v = *(const float4*)(x + (long long)s * FEAT + c);
  float* o = summed + (long long)d * FEAT + c;
  atomicAdd(o + 0, v.x);
  atomicAdd(o + 1, v.y);
  atomicAdd(o + 2, v.z);
  atomicAdd(o + 3, v.w);
}

// ------------------------------------------------------------- fused GEMM
// out[n, :] = act( (summed[n]*recip[n] | self[n]) @ Wt^T + b )
// 8 waves/block, 16 rows per wave, K = 256 in 8 tiles of 32 (bf16 WMMA).
template <int DOUT, bool RELU>
__global__ __launch_bounds__(256) void sage_gemm(
    const float* __restrict__ summed, const float* __restrict__ recip,
    const float* __restrict__ self,   const __bf16* __restrict__ Wt,
    const float* __restrict__ bias,   float* __restrict__ out, int n_nodes) {
  __shared__ __align__(32) __bf16 sW[DOUT * 256];  // only LDS object -> offset 0

  const int tid = threadIdx.x;
  const int wave = tid >> 5, lane = tid & 31;

#if __has_builtin(__builtin_amdgcn_tensor_load_to_lds) && \
    __has_builtin(__builtin_amdgcn_s_wait_tensorcnt)
  // Tensor Data Mover: one DMA brings the whole DOUT*256 bf16 weight tile
  // into LDS. D# per ISA 8.3/8.4: degenerate 2D tensor = 1 row of T0
  // 2-byte elements; groups 2/3 zero (tile_dim3=0 -> unused).
  if (wave == 0) {
    const unsigned long long ga = (unsigned long long)Wt;
    const unsigned T0 = (unsigned)(DOUT * 256);   // elements (2B each)
    v4u g0 = { 1u,                                 // count=1, is_restore=0
               0u,                                 // lds_addr = 0 (sW base)
               (unsigned)(ga & 0xFFFFFFFFu),
               (unsigned)((ga >> 32) & 0x01FFFFFFu) | (2u << 30) };  // type=2
    v8i g1 = { (int)0x00010000u,                       // data_size=1 (2 bytes)
               (int)((T0 & 0xFFFFu) << 16),            // tensor_dim0 lo
               (int)((T0 >> 16) | (1u << 16)),         // dim0 hi | tensor_dim1=1
               (int)((T0 & 0xFFFFu) << 16),            // tile_dim0 = T0
               1,                                      // tile_dim1=1, tile_dim2=0
               (int)T0,                                // tensor_dim0_stride lo
               0, 0 };
    v4i z4 = {0, 0, 0, 0};
#if defined(__clang_major__) && (__clang_major__ >= 23)
    v8i z8 = {0, 0, 0, 0, 0, 0, 0, 0};
    __builtin_amdgcn_tensor_load_to_lds(g0, g1, z4, z4, z8, 0);
#else
    __builtin_amdgcn_tensor_load_to_lds(g0, g1, z4, z4, 0);
#endif
    __builtin_amdgcn_s_wait_tensorcnt(0);
  }
#else
  {  // fallback: cooperative 16B weight staging into LDS
    const uint4* g = (const uint4*)Wt;
    uint4* l = (uint4*)sW;
    const int total = DOUT * 256 / 8;
    for (int i = tid; i < total; i += 256) l[i] = g[i];
  }
#endif
  __syncthreads();

  const int half = lane >> 4, lrow = lane & 15;
  const int rowBase = blockIdx.x * 128 + wave * 16;
  const int row = rowBase + lrow;
  const int rowc = row < n_nodes ? row : (n_nodes - 1);  // clamp, keep EXEC full
  const float rs = recip[rowc];
  const float* aggp  = summed + (long long)rowc * FEAT;
  const float* selfp = self   + (long long)rowc * FEAT;

  // A fragments: contiguous K per lane half (same order as B).
  // elem e in [0,8):  K = kt*32 + half*16 + e
  // elem e in [8,16): K = kt*32 + half*16 + e
  v16bf A[8];
#pragma unroll
  for (int kt = 0; kt < 8; ++kt) {
    const int kb0 = kt * 32 + half * 16;
    const int kb1 = kb0 + 8;
#pragma unroll
    for (int e2 = 0; e2 < 8; ++e2) {
      const int k0 = kb0 + e2;
      const int k1 = kb1 + e2;
      float a0 = (k0 < FEAT) ? aggp[k0] * rs : selfp[k0 - FEAT];
      float a1 = (k1 < FEAT) ? aggp[k1] * rs : selfp[k1 - FEAT];
      A[kt][e2]     = f2bf(a0);
      A[kt][e2 + 8] = f2bf(a1);
    }
  }

  const bool fullWave = (rowBase + 16) <= n_nodes;  // wave-uniform
  float* const outw = out + (long long)rowBase * DOUT;

#pragma unroll
  for (int nt = 0; nt < DOUT / 16; ++nt) {
    const int col = nt * 16 + lrow;
    // B fragment: col fixed per lane, K = kt*32 + half*16 + [0,16) contiguous.
    // Preload all 8 fragments, then chain the WMMAs back-to-back.
    const __bf16* wp = sW + col * 256 + half * 16;
    v16bf Bf[8];
#pragma unroll
    for (int kt = 0; kt < 8; ++kt) Bf[kt] = *(const v16bf*)(wp + kt * 32);

    v8f acc = {};
#pragma unroll
    for (int kt = 0; kt < 8; ++kt)
      acc = __builtin_amdgcn_wmma_f32_16x16x32_bf16(
          false, A[kt], false, Bf[kt], (short)0, acc, false, false);

    const float bb = bias[col];
    if (fullWave) {  // common case: no bounds checks, straight-line stores
#pragma unroll
      for (int r = 0; r < 8; ++r) {
        float v = acc[r] + bb;
        if (RELU) v = fmaxf(v, 0.0f);
        outw[(half * 8 + r) * DOUT + col] = v;  // D: VGPR r -> M = r + 8*half
      }
    } else {
#pragma unroll
      for (int r = 0; r < 8; ++r) {
        const int orow = rowBase + half * 8 + r;
        float v = acc[r] + bb;
        if (RELU) v = fmaxf(v, 0.0f);
        if (orow < n_nodes) out[(long long)orow * DOUT + col] = v;
      }
    }
  }
}

// ------------------------------------------------------------------ driver
extern "C" void kernel_launch(void* const* d_in, const int* in_sizes, int n_in,
                              void* d_out, int out_size, void* d_ws, size_t ws_size,
                              hipStream_t stream) {
  const float*     x   = (const float*)d_in[0];
  const long long* ei  = (const long long*)d_in[1];
  const float*     Wl0 = (const float*)d_in[2];
  const float*     Wr0 = (const float*)d_in[3];
  const float*     b0  = (const float*)d_in[4];
  const float*     Wl1 = (const float*)d_in[5];
  const float*     Wr1 = (const float*)d_in[6];
  const float*     b1  = (const float*)d_in[7];
  const float*     Wl2 = (const float*)d_in[8];
  const float*     Wr2 = (const float*)d_in[9];
  const float*     b2  = (const float*)d_in[10];

  const int n_nodes = in_sizes[0] / FEAT;  // 100000
  const int n_edges = in_sizes[1] / 2;     // 1600000
  const long long* src = ei;
  const long long* dst = ei + n_edges;

  // workspace carve-out (256B aligned)
  char* ws = (char*)d_ws;
  size_t off = 0;
  auto carve = [&](size_t bytes) {
    void* p = ws + off;
    off += (bytes + 255) & ~(size_t)255;
    return p;
  };
  const size_t featBytes = (size_t)n_nodes * FEAT * sizeof(float);
  float*  recip  = (float*)carve((size_t)n_nodes * sizeof(float));
  float*  summed = (float*)carve(featBytes);
  float*  h0     = (float*)carve(featBytes);
  float*  h1     = (float*)carve(featBytes);
  __bf16* Wt0    = (__bf16*)carve(128 * 256 * sizeof(__bf16));
  __bf16* Wt1    = (__bf16*)carve(128 * 256 * sizeof(__bf16));
  __bf16* Wt2    = (__bf16*)carve(64 * 256 * sizeof(__bf16));

  const int B = 256;
  const long long nfeat4 = (long long)n_nodes * FEAT / 4;
  const long long chunks = (long long)n_edges * 32;
  const int edgeGrid  = (n_edges + B - 1) / B;
  const int chunkGrid = (int)((chunks + B - 1) / B);
  const int gemmGrid  = (n_nodes + 127) / 128;

  // degrees -> reciprocal (shared by all layers)
  zero_f32<<<1024, B, 0, stream>>>(recip, n_nodes / 4);
  deg_count<<<edgeGrid, B, 0, stream>>>(dst, recip, n_edges);
  deg_recip<<<(n_nodes + B - 1) / B, B, 0, stream>>>(recip, n_nodes);

  // weight packing (fp32 -> bf16, transposed [col][k])
  prep_w<<<(128 * 256 + B - 1) / B, B, 0, stream>>>(Wl0, Wr0, Wt0, 128);
  prep_w<<<(128 * 256 + B - 1) / B, B, 0, stream>>>(Wl1, Wr1, Wt1, 128);
  prep_w<<<(64 * 256 + B - 1) / B, B, 0, stream>>>(Wl2, Wr2, Wt2, 64);

  // layer 0: x -> h0
  zero_f32<<<2048, B, 0, stream>>>(summed, nfeat4);
  scatter_feat<<<chunkGrid, B, 0, stream>>>(x, src, dst, summed, n_edges);
  sage_gemm<128, true><<<gemmGrid, B, 0, stream>>>(summed, recip, x, Wt0, b0, h0, n_nodes);

  // layer 1: h0 -> h1
  zero_f32<<<2048, B, 0, stream>>>(summed, nfeat4);
  scatter_feat<<<chunkGrid, B, 0, stream>>>(h0, src, dst, summed, n_edges);
  sage_gemm<128, true><<<gemmGrid, B, 0, stream>>>(summed, recip, h0, Wt1, b1, h1, n_nodes);

  // layer 2: h1 -> out (no ReLU)
  zero_f32<<<2048, B, 0, stream>>>(summed, nfeat4);
  scatter_feat<<<chunkGrid, B, 0, stream>>>(h1, src, dst, summed, n_edges);
  sage_gemm<64, false><<<gemmGrid, B, 0, stream>>>(summed, recip, h1, Wt2, b2,
                                                   (float*)d_out, n_nodes);
}
